// HierarchicalFeatureAggregator_28956669510180
// MI455X (gfx1250) — matrix operand order repaired and checked by reading, not compiled, not run on hardware
//
#include <hip/hip_runtime.h>

// ---------------------------------------------------------------------------
// HierarchicalFeatureAggregator on gfx1250 (MI455X), WMMA f16 pipeline.
// B = 32768 rows, H = 256. All GEMMs use v_wmma_f32_16x16x32_f16 with f32
// accumulation; LayerNorm / sigmoid-gate epilogues fused via lane butterflies.
// NOTE: the inner loop is deliberately "naive" — the compiler software-
// pipelines the strided 16-bit B-fragment loads itself;
// manual double-buffering regressed scheduling under register pressure.
// ---------------------------------------------------------------------------

typedef __attribute__((ext_vector_type(16))) _Float16     v16h;
typedef __attribute__((ext_vector_type(8)))  float        v8f;
typedef __attribute__((ext_vector_type(4)))  float        v4f;
typedef __attribute__((ext_vector_type(4)))  _Float16     v4h;
typedef __attribute__((ext_vector_type(4)))  unsigned int v4u;

union AFrag { v16h h; unsigned int u[8]; };
union BFrag { v16h h; v4u q[2]; };

#define BQ 32768   // batch rows
#define HD 256     // hidden dim

// ---------------------------------------------------------------------------
// Core WMMA strip: 16 rows x 256 cols, K = kSteps*32.
// A(row,k) = A + seg[k>>8] + row*lda + (k&255); W is [K,256] f16 row-major.
// ---------------------------------------------------------------------------
__device__ __forceinline__
void wmma_strip(const _Float16* __restrict__ A, long s0, long s1, long s2,
                long row, int lda, const _Float16* __restrict__ W,
                int kSteps, int half, int lane, v8f acc[16])
{
    const long seg[3] = { s0, s1, s2 };
    for (int ks = 0; ks < kSteps; ++ks) {
        const int kbase = ks << 5;
        AFrag a;
#pragma unroll
        for (int v = 0; v < 8; ++v) {   // ISA 16-bit A layout: lane&15 = M row
            const int k = kbase + ((v >> 2) << 4) + (half << 3) + ((v & 3) << 1);
            a.u[v] = *(const unsigned int*)(A + seg[k >> 8] + row * (long)lda + (k & 255));
        }
        const _Float16* wcol = W + (lane & 15);  // lane&15 = N column of B
#pragma unroll
        for (int nt = 0; nt < 16; ++nt) {
            BFrag b;
#pragma unroll
            for (int s = 0; s < 16; ++s)
                b.h[s] = wcol[(long)(kbase + ((s >> 3) << 4) + (half << 3) + (s & 7)) * HD + nt * 16];
            acc[nt] = __builtin_amdgcn_wmma_f32_16x16x32_f16(
                false, a.h, false, b.h, (short)0, acc[nt], false, false);
        }
    }
}

// ---------------------------------------------------------------------------
__global__ __launch_bounds__(256)
void cvt_f32_f16_kernel(const float* __restrict__ s, _Float16* __restrict__ d, long n) {
    const long i = ((long)blockIdx.x * blockDim.x + threadIdx.x) * 4;
    if (i >= n) return;                       // n is a multiple of 4
    const v4f x = *(const v4f*)(s + i);
    v4h y;
#pragma unroll
    for (int j = 0; j < 4; ++j) y[j] = (_Float16)x[j];
    *(v4h*)(d + i) = y;
}

// ---------------------------------------------------------------------------
// Generic GEMM: C[row, n] = epilogue( sum_k A(row,k) * W[k,n] + bias[n] )
// mode 0: optional ReLU, store f32 and/or f16
// mode 1: LayerNorm(g,b) over n=0..255, store f32 and/or f16
// Wave = 16 rows x 256 cols (16 n-tiles). Block = 8 waves = 128 rows.
// ---------------------------------------------------------------------------
__global__ __launch_bounds__(256)
void gemm_kernel(const _Float16* __restrict__ A, long s0, long s1, long s2,
                 int lda, int K,
                 const _Float16* __restrict__ W, const float* __restrict__ bias,
                 int mode, int relu,
                 float* __restrict__ out32, _Float16* __restrict__ out16,
                 const float* __restrict__ lnG, const float* __restrict__ lnB)
{
    const int lane = threadIdx.x & 31;
    const int wave = threadIdx.x >> 5;
    const int half = lane >> 4;
    const int m    = lane & 15;
    const int rowBase = blockIdx.x * 128 + wave * 16;
    const long row = rowBase + m;

    v8f acc[16];
#pragma unroll
    for (int nt = 0; nt < 16; ++nt)
#pragma unroll
        for (int v = 0; v < 8; ++v) acc[nt][v] = 0.0f;

    wmma_strip(A, s0, s1, s2, row, lda, W, K >> 5, half, lane, acc);

    if (mode == 0) {                    // bias (+ReLU) store
#pragma unroll
        for (int nt = 0; nt < 16; ++nt) {
            const int n = nt * 16 + m;
            const float bn = bias[n];
#pragma unroll
            for (int v = 0; v < 8; ++v) {
                const long r = rowBase + v + half * 8;   // C layout: row = v + 8*half
                float val = acc[nt][v] + bn;
                if (relu) val = fmaxf(val, 0.0f);
                if (out32) out32[r * HD + n] = val;
                if (out16) out16[r * HD + n] = (_Float16)val;
            }
        }
    } else {                            // bias + LayerNorm store
        float sx[8], sxx[8];
#pragma unroll
        for (int v = 0; v < 8; ++v) { sx[v] = 0.0f; sxx[v] = 0.0f; }
#pragma unroll
        for (int nt = 0; nt < 16; ++nt) {
            const float bn = bias[nt * 16 + m];
#pragma unroll
            for (int v = 0; v < 8; ++v) {
                const float val = acc[nt][v] + bn;
                acc[nt][v] = val;
                sx[v] += val; sxx[v] += val * val;
            }
        }
#pragma unroll
        for (int off = 1; off < 16; off <<= 1) {     // row spread over 16 lanes
#pragma unroll
            for (int v = 0; v < 8; ++v) {
                sx[v]  += __shfl_xor(sx[v],  off, 32);
                sxx[v] += __shfl_xor(sxx[v], off, 32);
            }
        }
        float mean[8], rstd[8];
#pragma unroll
        for (int v = 0; v < 8; ++v) {
            mean[v] = sx[v] * (1.0f / 256.0f);
            const float var = sxx[v] * (1.0f / 256.0f) - mean[v] * mean[v];
            rstd[v] = rsqrtf(var + 1e-5f);
        }
#pragma unroll
        for (int nt = 0; nt < 16; ++nt) {
            const int n = nt * 16 + m;
            const float g = lnG[n], bb = lnB[n];
#pragma unroll
            for (int v = 0; v < 8; ++v) {
                const long r = rowBase + v + half * 8;
                const float y = (acc[nt][v] - mean[v]) * rstd[v] * g + bb;
                if (out32) out32[r * HD + n] = y;
                if (out16) out16[r * HD + n] = (_Float16)y;
            }
        }
    }
}

// ---------------------------------------------------------------------------
// Routing MLP, fully fused per directed pair p (grid.z):
//   h = relu([feats[src] || feats[tgt]] @ rW1[p] + rb1[p])   (K=512, N=256)
//   w = sigmoid(h . rW2[p] + rb2[p])                          (per row scalar)
// h never leaves registers: each n-tile is ReLU'd, scaled by rW2, reduced.
// ---------------------------------------------------------------------------
__global__ __launch_bounds__(256)
void route_kernel(const _Float16* __restrict__ feats16, long levStride,
                  const _Float16* __restrict__ rW1, const float* __restrict__ rb1,
                  const float* __restrict__ rW2, const float* __restrict__ rb2,
                  float* __restrict__ wout)
{
    const int p = blockIdx.z;
    const int SRCp[6] = {0, 0, 1, 1, 2, 2};
    const int TGTp[6] = {1, 2, 0, 2, 0, 1};
    const long s0 = SRCp[p] * levStride;
    const long s1 = TGTp[p] * levStride;
    const _Float16* W   = rW1 + (long)p * 512 * HD;
    const float* bias   = rb1 + p * HD;
    const float* w2     = rW2 + p * HD;
    const float  b2     = rb2[p];

    const int lane = threadIdx.x & 31;
    const int wave = threadIdx.x >> 5;
    const int half = lane >> 4;
    const int m    = lane & 15;
    const int rowBase = blockIdx.x * 128 + wave * 16;
    const long row = rowBase + m;

    v8f acc[16];
#pragma unroll
    for (int nt = 0; nt < 16; ++nt)
#pragma unroll
        for (int v = 0; v < 8; ++v) acc[nt][v] = 0.0f;

    wmma_strip(feats16, s0, s1, 0, row, HD, W, 16 /* K=512 */, half, lane, acc);

    float part[8];
#pragma unroll
    for (int v = 0; v < 8; ++v) part[v] = 0.0f;
#pragma unroll
    for (int nt = 0; nt < 16; ++nt) {
        const int n = nt * 16 + m;
        const float bn = bias[n], wn = w2[n];
#pragma unroll
        for (int v = 0; v < 8; ++v)
            part[v] += fmaxf(acc[nt][v] + bn, 0.0f) * wn;
    }
#pragma unroll
    for (int off = 1; off < 16; off <<= 1)
#pragma unroll
        for (int v = 0; v < 8; ++v)
            part[v] += __shfl_xor(part[v], off, 32);

    if (m == 0) {
#pragma unroll
        for (int v = 0; v < 8; ++v) {
            const long r = rowBase + v + half * 8;
            const float s = part[v] + b2;
            wout[(long)p * BQ + r] = 1.0f / (1.0f + __expf(-s));
        }
    }
}

// ---------------------------------------------------------------------------
// upd[l] = feats[l] + sum_{p: TGT[p]==l} w[p] * feats[SRC[p]]  -> f16
// TGT=[1,2,0,2,0,1]: l0 <- p2(s1),p4(s2); l1 <- p0(s0),p5(s2); l2 <- p1(s0),p3(s1)
// grid: (L/(256*4), 3); 4 elements per thread, vectorized b128/b64.
// ---------------------------------------------------------------------------
__global__ __launch_bounds__(256)
void update_kernel(const float* __restrict__ feats32, const float* __restrict__ w,
                   _Float16* __restrict__ upd16)
{
    const long L = (long)BQ * HD;
    const int  lvl = blockIdx.y;
    const long idx = ((long)blockIdx.x * blockDim.x + threadIdx.x) * 4;  // within level
    if (idx >= L) return;
    const long b = idx >> 8;
    const int PA[3] = {2, 0, 1}, SA[3] = {1, 0, 0};
    const int PB[3] = {4, 5, 3}, SB[3] = {2, 2, 1};
    const float wa = w[(long)PA[lvl] * BQ + b];
    const float wb = w[(long)PB[lvl] * BQ + b];
    const v4f f  = *(const v4f*)(feats32 + (long)lvl     * L + idx);
    const v4f fa = *(const v4f*)(feats32 + (long)SA[lvl] * L + idx);
    const v4f fb = *(const v4f*)(feats32 + (long)SB[lvl] * L + idx);
    v4h o;
#pragma unroll
    for (int j = 0; j < 4; ++j) o[j] = (_Float16)(f[j] + wa * fa[j] + wb * fb[j]);
    *(v4h*)(upd16 + (long)lvl * L + idx) = o;
}

// ---------------------------------------------------------------------------
extern "C" void kernel_launch(void* const* d_in, const int* in_sizes, int n_in,
                              void* d_out, int out_size, void* d_ws, size_t ws_size,
                              hipStream_t stream) {
    (void)in_sizes; (void)n_in; (void)out_size; (void)ws_size;
    const int B = BQ;
    const long L = (long)B * HD;           // elements per level feature tensor

    const float* x_fine   = (const float*)d_in[0];
    const float* x_medium = (const float*)d_in[1];
    const float* x_coarse = (const float*)d_in[2];
    const float* pW0 = (const float*)d_in[3];  const float* pb0 = (const float*)d_in[4];
    const float* pW1 = (const float*)d_in[5];  const float* pb1 = (const float*)d_in[6];
    const float* pW2 = (const float*)d_in[7];  const float* pb2 = (const float*)d_in[8];
    const float* rW1 = (const float*)d_in[9];  const float* rb1 = (const float*)d_in[10];
    const float* rW2 = (const float*)d_in[11]; const float* rb2 = (const float*)d_in[12];
    const float* lpW1 = (const float*)d_in[13]; const float* lpb1 = (const float*)d_in[14];
    const float* lpW2 = (const float*)d_in[15]; const float* lpb2 = (const float*)d_in[16];
    const float* lng = (const float*)d_in[17];  const float* lnb = (const float*)d_in[18];
    const float* faW1 = (const float*)d_in[19]; const float* fab1 = (const float*)d_in[20];
    const float* faW2 = (const float*)d_in[21]; const float* fab2 = (const float*)d_in[22];
    const float* fng = (const float*)d_in[23];  const float* fnb = (const float*)d_in[24];

    // -------- workspace carve-out (256B aligned regions) --------
    char* ws = (char*)d_ws;
    size_t off = 0;
    auto alloc = [&](size_t bytes) -> void* {
        off = (off + 255) & ~(size_t)255;
        void* p = ws + off; off += bytes; return p;
    };
    _Float16* x16     = (_Float16*)alloc((size_t)B * (128 + 256 + 512) * 2); // also h2
    float*    feats32 = (float*)   alloc((size_t)3 * L * 4);
    _Float16* feats16 = (_Float16*)alloc((size_t)3 * L * 2);
    _Float16* upd16   = (_Float16*)alloc((size_t)3 * L * 2);                 // also h4
    _Float16* w16     = (_Float16*)alloc((size_t)1671168 * 2);
    float*    wbuf    = (float*)   alloc((size_t)6 * B * 4);

    _Float16* x16f = x16;
    _Float16* x16m = x16 + (long)B * 128;
    _Float16* x16c = x16 + (long)B * 384;
    _Float16* h2   = x16;     // alias (x16 only needed for projections)
    _Float16* h4   = upd16;   // alias (upd16 dead after last L1 GEMM)

    _Float16* pW0_16  = w16;
    _Float16* pW1_16  = pW0_16 + 32768;
    _Float16* pW2_16  = pW1_16 + 65536;
    _Float16* rW1_16  = pW2_16 + 131072;
    _Float16* lpW1_16 = rW1_16 + 786432;
    _Float16* lpW2_16 = lpW1_16 + 196608;
    _Float16* faW1_16 = lpW2_16 + 196608;
    _Float16* faW2_16 = faW1_16 + 196608;

    auto cvt = [&](const float* s, _Float16* d, long n) {
        cvt_f32_f16_kernel<<<(int)((n / 4 + 255) / 256), 256, 0, stream>>>(s, d, n);
    };
    // inputs -> f16
    cvt(x_fine,   x16f, (long)B * 128);
    cvt(x_medium, x16m, (long)B * 256);
    cvt(x_coarse, x16c, (long)B * 512);
    // weights -> f16
    cvt(pW0,  pW0_16,  32768);
    cvt(pW1,  pW1_16,  65536);
    cvt(pW2,  pW2_16,  131072);
    cvt(rW1,  rW1_16,  786432);
    cvt(lpW1, lpW1_16, 196608);
    cvt(lpW2, lpW2_16, 196608);
    cvt(faW1, faW1_16, 196608);
    cvt(faW2, faW2_16, 65536);

    const dim3 gGrid(B / 128, 1, 1);
    const dim3 blk(256, 1, 1);
    auto gemm = [&](const _Float16* A, long s0, long s1, long s2, int lda, int K,
                    const _Float16* W, const float* bias, int mode, int relu,
                    float* o32, _Float16* o16, const float* g, const float* bv) {
        gemm_kernel<<<gGrid, blk, 0, stream>>>(A, s0, s1, s2, lda, K, W, bias,
                                               mode, relu, o32, o16, g, bv);
    };

    // -------- projections: feats[l] = x_l @ pW_l + pb_l --------
    gemm(x16f, 0, 256, 512, 128, 128, pW0_16, pb0, 0, 0,
         feats32,         feats16,         nullptr, nullptr);
    gemm(x16m, 0, 256, 512, 256, 256, pW1_16, pb1, 0, 0,
         feats32 + L,     feats16 + L,     nullptr, nullptr);
    gemm(x16c, 0, 256, 512, 512, 512, pW2_16, pb2, 0, 0,
         feats32 + 2 * L, feats16 + 2 * L, nullptr, nullptr);

    // -------- 3 routing iterations --------
    for (int it = 0; it < 3; ++it) {
        float* wPtr = (it == 2) ? ((float*)d_out + L) : wbuf;  // last iter: w slice of output

        route_kernel<<<dim3(B / 128, 1, 6), blk, 0, stream>>>(
            feats16, L, rW1_16, rb1, rW2, rb2, wPtr);

        update_kernel<<<dim3((int)(L / (256 * 4)), 3, 1), blk, 0, stream>>>(
            feats32, wPtr, upd16);

        for (int l = 0; l < 3; ++l) {
            // h2 = relu(upd @ lpW1[l] + lpb1[l])
            gemm(upd16 + l * L, 0, 256, 512, 256, 256,
                 lpW1_16 + l * 65536, lpb1 + l * 256, 0, 1,
                 nullptr, h2 + l * L, nullptr, nullptr);
            // feats = LN(h2 @ lpW2[l] + lpb2[l]; lng[l], lnb[l])
            gemm(h2 + l * L, 0, 256, 512, 256, 256,
                 lpW2_16 + l * 65536, lpb2 + l * 256, 1, 0,
                 feats32 + l * L, feats16 + l * L, lng + l * 256, lnb + l * 256);
        }
    }

    // -------- final aggregator --------
    // h4 = relu([f0||f1||f2] @ faW1 + fab1)   (K = 768, segmented A)
    gemm(feats16, 0, L, 2 * L, 256, 768, faW1_16, fab1, 0, 1,
         nullptr, h4, nullptr, nullptr);
    // out = LN(h4 @ faW2 + fab2; fng, fnb)
    gemm(h4, 0, 256, 512, 256, 256, faW2_16, fab2, 1, 0,
         (float*)d_out, nullptr, fng, fnb);
}
